// _SSMCore_12369505812974
// MI455X (gfx1250) — matrix operand order repaired and checked by reading, not compiled, not run on hardware
//
#include <hip/hip_runtime.h>
#include <hip/hip_bf16.h>
#include <math.h>

// ---------------- Problem constants (from reference) ----------------
#define BATCH   2
#define LSEQ    1024
#define DMODEL  1024
#define DSTATE  16
#define DCONV   4
#define DINNER  2048            // DMODEL * EXPAND
#define ROWS    (BATCH * LSEQ)  // 2048 token rows
#define XZCOLS  (2 * DINNER)    // 4096
#define BCDT_STRIDE 48          // padded row stride for 33-wide bcdt

// ---------------- WMMA vector types (probe-confirmed signatures) ----
typedef __attribute__((ext_vector_type(16))) __bf16 v16bf;
typedef __attribute__((ext_vector_type(8)))  __bf16 v8bf;
typedef __attribute__((ext_vector_type(8)))  float  v8f;

union Frag16 { v16bf v; v8bf h[2]; };

// ---------------- fp32 -> bf16 convert (A-side, row-major kept) -----
__global__ __launch_bounds__(256) void cvt_f32_to_bf16(const float* __restrict__ src,
                                                       __bf16* __restrict__ dst,
                                                       int n) {
    int i = blockIdx.x * blockDim.x + threadIdx.x;
    if (i < n) dst[i] = (__bf16)src[i];
}

// ---------------- fp32 -> bf16 convert + pack into B-fragment layout -
// Packed layout: Bp[(((nTile*KB + kb)*32 + lane)*16 + e],
//   where lane = hi*16 + n, element e maps to B[kb*32 + hi*16 + e][nTile*16 + n].
// One wave's k-step fragment is then 32 lanes x 32 contiguous bytes.
__global__ __launch_bounds__(256) void pack_b_bf16(const float* __restrict__ B,
                                                   __bf16* __restrict__ Bp,
                                                   int K, int N) {
    int idx = blockIdx.x * blockDim.x + threadIdx.x;   // over K*N
    if (idx >= K * N) return;
    const int KB    = K >> 5;
    int e     = idx & 15;
    int lane  = (idx >> 4) & 31;
    int blk   = idx >> 9;            // nTile*KB + kb
    int kb    = blk % KB;
    int nTile = blk / KB;
    int hi = lane >> 4, n = lane & 15;
    int k  = kb * 32 + hi * 16 + e;
    Bp[idx] = (__bf16)B[(size_t)k * N + nTile * 16 + n];
}

// ---------------- WMMA bf16 GEMM: C[MxN] = A[MxK] * B[KxN] ----------
// A row-major bf16; B pre-packed (pack_b_bf16). One wave computes a
// 16x64 output strip: one A fragment feeds 4 WMMA accumulators.
// Fragment k order is contiguous (k = hi*16 + e) for both A and B (wave32).
__global__ __launch_bounds__(256) void wmma_gemm_bf16(const __bf16* __restrict__ A,
                                                      const __bf16* __restrict__ Bp,
                                                      float* __restrict__ C,
                                                      int M, int N, int K) {
    const int lane    = threadIdx.x & 31;
    const int waveId  = blockIdx.x * (blockDim.x >> 5) + (threadIdx.x >> 5);
    const int nGroups = N >> 6;                   // groups of 4 N-tiles
    const int total   = (M >> 4) * nGroups;
    if (waveId >= total) return;                  // wave-uniform branch
    const int tileM  = (waveId / nGroups) << 4;
    const int nGroup =  waveId % nGroups;
    const int m  = lane & 15;
    const int hi = lane >> 4;
    const int KB = K >> 5;

    v8f acc0 = {}, acc1 = {}, acc2 = {}, acc3 = {};
    const __bf16* Arow = A + (size_t)(tileM + m) * K;
    // per-lane packed-B pointers for the 4 N-tiles of this group
    const __bf16* bp0 = Bp + (((size_t)(nGroup * 4 + 0) * KB) * 32 + lane) * 16;
    const __bf16* bp1 = Bp + (((size_t)(nGroup * 4 + 1) * KB) * 32 + lane) * 16;
    const __bf16* bp2 = Bp + (((size_t)(nGroup * 4 + 2) * KB) * 32 + lane) * 16;
    const __bf16* bp3 = Bp + (((size_t)(nGroup * 4 + 3) * KB) * 32 + lane) * 16;

    for (int kb = 0; kb < KB; ++kb) {
        Frag16 a, b0, b1, b2, b3;
        // A fragment: two contiguous 16B loads (K = hi*16 .. hi*16+15)
        a.h[0] = *reinterpret_cast<const v8bf*>(Arow + hi * 16);
        a.h[1] = *reinterpret_cast<const v8bf*>(Arow + hi * 16 + 8);
        // B fragments: 32 contiguous bytes per lane each
        b0.h[0] = *reinterpret_cast<const v8bf*>(bp0);
        b0.h[1] = *reinterpret_cast<const v8bf*>(bp0 + 8);
        b1.h[0] = *reinterpret_cast<const v8bf*>(bp1);
        b1.h[1] = *reinterpret_cast<const v8bf*>(bp1 + 8);
        b2.h[0] = *reinterpret_cast<const v8bf*>(bp2);
        b2.h[1] = *reinterpret_cast<const v8bf*>(bp2 + 8);
        b3.h[0] = *reinterpret_cast<const v8bf*>(bp3);
        b3.h[1] = *reinterpret_cast<const v8bf*>(bp3 + 8);
        // stream-ahead hint for the A row (global_prefetch_b8)
        __builtin_prefetch(Arow + 256, 0, 1);
        acc0 = __builtin_amdgcn_wmma_f32_16x16x32_bf16(false, a.v, false, b0.v,
                                                       (short)0, acc0, false, false);
        acc1 = __builtin_amdgcn_wmma_f32_16x16x32_bf16(false, a.v, false, b1.v,
                                                       (short)0, acc1, false, false);
        acc2 = __builtin_amdgcn_wmma_f32_16x16x32_bf16(false, a.v, false, b2.v,
                                                       (short)0, acc2, false, false);
        acc3 = __builtin_amdgcn_wmma_f32_16x16x32_bf16(false, a.v, false, b3.v,
                                                       (short)0, acc3, false, false);
        Arow += 32;
        bp0 += 512; bp1 += 512; bp2 += 512; bp3 += 512;   // 32 lanes * 16 elems
    }
    // D layout: vgpr r -> M = hi*8 + r, N = lane&15
    #pragma unroll
    for (int r = 0; r < 8; ++r) {
        size_t base = (size_t)(tileM + hi * 8 + r) * N + (nGroup << 6) + m;
        C[base]      = acc0[r];
        C[base + 16] = acc1[r];
        C[base + 32] = acc2[r];
        C[base + 48] = acc3[r];
    }
}

// ---------------- depthwise causal conv (k=4) + bias + SiLU ---------
__global__ __launch_bounds__(256) void conv_silu_kernel(const float* __restrict__ xz,
                                                        const float* __restrict__ conv_w,
                                                        const float* __restrict__ conv_b,
                                                        float* __restrict__ xc) {
    int idx = blockIdx.x * blockDim.x + threadIdx.x;           // over B*L*DINNER
    if (idx >= ROWS * DINNER) return;
    int c   = idx & (DINNER - 1);
    int row = idx >> 11;                                       // b*L + l
    int l   = row & (LSEQ - 1);
    int b   = row >> 10;
    float s = conv_b[c];
    #pragma unroll
    for (int j = 0; j < DCONV; ++j) {
        int lj = l - (DCONV - 1) + j;
        if (lj >= 0)
            s = fmaf(conv_w[c * DCONV + j],
                     xz[(size_t)(b * LSEQ + lj) * XZCOLS + c], s);
    }
    float sig = 1.0f / (1.0f + expf(-s));
    xc[idx] = s * sig;
}

// ---------------- x_proj: bcdt[row][0..32] = xc_row . x_proj_w ------
__global__ __launch_bounds__(256) void xproj_kernel(const float* __restrict__ xc,
                                                    const float* __restrict__ xw,
                                                    float* __restrict__ bcdt) {
    int idx = blockIdx.x * blockDim.x + threadIdx.x;           // over ROWS*33
    if (idx >= ROWS * 33) return;
    int row = idx / 33, col = idx - row * 33;
    const float* xr = xc + (size_t)row * DINNER;
    float s = 0.0f;
    for (int k = 0; k < DINNER; ++k)
        s = fmaf(xr[k], xw[(size_t)k * 33 + col], s);
    bcdt[(size_t)row * BCDT_STRIDE + col] = s;
}

// ---------------- selective scan: one thread per (b, channel) -------
// h[16] lives in registers; 1024 sequential steps; fuses softplus(dt),
// dA=exp(dt*A), clip, y += h*C, D-residual, SiLU(z) gating; emits bf16.
__global__ __launch_bounds__(32) void ssm_scan_kernel(const float* __restrict__ xz,
                                                      const float* __restrict__ xc,
                                                      const float* __restrict__ bcdt,
                                                      const float* __restrict__ dt_w,
                                                      const float* __restrict__ dt_b,
                                                      const float* __restrict__ A_log,
                                                      const float* __restrict__ Dp,
                                                      __bf16* __restrict__ yBF) {
    // 128 single-wave workgroups: blk = b*64 + (c>>5)
    int blk = blockIdx.x;
    int b   = blk >> 6;
    int c   = ((blk & 63) << 5) + threadIdx.x;                 // channel in [0,2048)
    float Afac[DSTATE];
    #pragma unroll
    for (int s = 0; s < DSTATE; ++s) {
        float al = A_log[c * DSTATE + s];
        al = fminf(fmaxf(al, -6.0f), 6.0f);
        Afac[s] = -expf(al);
    }
    const float dtw = dt_w[c], dtb = dt_b[c], dres = Dp[c];
    float h[DSTATE];
    #pragma unroll
    for (int s = 0; s < DSTATE; ++s) h[s] = 0.0f;

    for (int t = 0; t < LSEQ; ++t) {
        const size_t row = (size_t)(b * LSEQ + t);
        const float* bc = bcdt + row * BCDT_STRIDE;            // L2-broadcast reads
        float arg = fmaf(bc[0], dtw, dtb);
        float dt  = (arg > 20.0f) ? arg : log1pf(expf(arg));   // softplus
        float xct = xc[row * DINNER + c];
        float y = 0.0f;
        #pragma unroll
        for (int s = 0; s < DSTATE; ++s) {
            float dA = expf(dt * Afac[s]);
            float hv = fmaf(dA, h[s], dt * bc[1 + s] * xct);
            hv = fminf(fmaxf(hv, -10000.0f), 10000.0f);
            h[s] = hv;
            y = fmaf(hv, bc[1 + DSTATE + s], y);
        }
        y = fmaf(xct, dres, y);
        float z = xz[row * XZCOLS + DINNER + c];
        y *= z / (1.0f + expf(-z));                            // * silu(z)
        yBF[row * DINNER + c] = (__bf16)y;
    }
}

// ---------------- host-side orchestration ---------------------------
extern "C" void kernel_launch(void* const* d_in, const int* in_sizes, int n_in,
                              void* d_out, int out_size, void* d_ws, size_t ws_size,
                              hipStream_t stream) {
    const float* x          = (const float*)d_in[0];
    const float* in_proj_w  = (const float*)d_in[1];
    const float* conv_w     = (const float*)d_in[2];
    const float* conv_b     = (const float*)d_in[3];
    const float* x_proj_w   = (const float*)d_in[4];
    const float* dt_w       = (const float*)d_in[5];
    const float* dt_b       = (const float*)d_in[6];
    const float* A_log      = (const float*)d_in[7];
    const float* Dp         = (const float*)d_in[8];
    const float* out_proj_w = (const float*)d_in[9];
    float* out = (float*)d_out;

    // workspace layout (bytes)
    char* ws = (char*)d_ws;
    float*  xz    = (float*)(ws);                                   // 2048*4096 f32 = 32 MB
    float*  xc    = (float*)(ws + (size_t)ROWS * XZCOLS * 4);       // 2048*2048 f32 = 16 MB
    float*  bcdt  = (float*)((char*)xc + (size_t)ROWS * DINNER * 4);// 2048*48  f32
    __bf16* xBF   = (__bf16*)((char*)bcdt + (size_t)ROWS * BCDT_STRIDE * 4); // 2048*1024
    __bf16* wInBF = xBF   + (size_t)ROWS * DMODEL;                  // 1024*4096 (packed)
    __bf16* wOutBF= wInBF + (size_t)DMODEL * XZCOLS;                // 2048*1024 (packed)
    __bf16* yBF   = wOutBF+ (size_t)DINNER * DMODEL;                // 2048*2048

    // 1) convert activations; convert+pack weights into B-fragment layout
    {
        int n = ROWS * DMODEL;
        cvt_f32_to_bf16<<<(n + 255) / 256, 256, 0, stream>>>(x, xBF, n);
        n = DMODEL * XZCOLS;
        pack_b_bf16<<<(n + 255) / 256, 256, 0, stream>>>(in_proj_w, wInBF,
                                                         DMODEL, XZCOLS);
        n = DINNER * DMODEL;
        pack_b_bf16<<<(n + 255) / 256, 256, 0, stream>>>(out_proj_w, wOutBF,
                                                         DINNER, DMODEL);
    }
    // 2) in_proj GEMM: [2048x1024] x [1024x4096] -> xz
    {
        int waves = (ROWS / 16) * (XZCOLS / 64);       // 8192 waves (16x64 strips)
        wmma_gemm_bf16<<<waves / 8, 256, 0, stream>>>(xBF, wInBF, xz,
                                                      ROWS, XZCOLS, DMODEL);
    }
    // 3) depthwise conv + SiLU
    {
        int n = ROWS * DINNER;
        conv_silu_kernel<<<(n + 255) / 256, 256, 0, stream>>>(xz, conv_w, conv_b, xc);
    }
    // 4) x_proj (N=33, scalar dot)
    {
        int n = ROWS * 33;
        xproj_kernel<<<(n + 255) / 256, 256, 0, stream>>>(xc, x_proj_w, bcdt);
    }
    // 5) selective scan (128 single-wave workgroups)
    ssm_scan_kernel<<<BATCH * (DINNER / 32), 32, 0, stream>>>(
        xz, xc, bcdt, dt_w, dt_b, A_log, Dp, yBF);
    // 6) out_proj GEMM: [2048x2048] x [2048x1024] -> d_out
    {
        int waves = (ROWS / 16) * (DMODEL / 64);       // 2048 waves
        wmma_gemm_bf16<<<waves / 8, 256, 0, stream>>>(yBF, wOutBF, out,
                                                      ROWS, DMODEL, DINNER);
    }
}
